// SturtGAT_48524540510791
// MI455X (gfx1250) — matrix operand order repaired and checked by reading, not compiled, not run on hardware
//
#include <hip/hip_runtime.h>
#include <hip/hip_bf16.h>

typedef __attribute__((ext_vector_type(16))) _Float16 v16h;
typedef __attribute__((ext_vector_type(8)))  _Float16 v8h;
typedef __attribute__((ext_vector_type(8)))  float    v8f;

#define N_NODES 50000
#define F_IN    500
#define KP0     512   // padded K for layer-0 GEMM
#define C0      64
#define HC0     128
#define C1      40
#define HC1     80
#define KP1     128
#define NEDGE   800000
#define ETOT    (NEDGE + N_NODES)
#define MTILES  (N_NODES / 16)   // 3125 exactly

__device__ __forceinline__ float lrelu_f(float x) { return x >= 0.f ? x : 0.2f * x; }
__device__ __forceinline__ float elu_f(float x)   { return x > 0.f ? x : __expf(x) - 1.f; }
// order-preserving f32 <-> u32 for atomic max
__device__ __forceinline__ unsigned fenc(float f) {
    unsigned u = __float_as_uint(f);
    return (u >> 31) ? ~u : (u | 0x80000000u);
}
__device__ __forceinline__ float fdec(unsigned e) {
    return (e >> 31) ? __uint_as_float(e & 0x7FFFFFFFu) : __uint_as_float(~e);
}

// ---------------- conversion / staging ----------------
__global__ void k_conv_x(const float* __restrict__ x, _Float16* __restrict__ xh) {
    long long gid = (long long)blockIdx.x * blockDim.x + threadIdx.x;
    if (gid >= (long long)N_NODES * KP0) return;
    int n = (int)(gid / KP0), k = (int)(gid % KP0);
    xh[gid] = (k < F_IN) ? (_Float16)x[(long long)n * F_IN + k] : (_Float16)0.f;
}

// Wt[c*KP + k] = W[k*Ncols + c]  (zero-padded K)  -> B fragment is contiguous
__global__ void k_conv_wt(const float* __restrict__ W, _Float16* __restrict__ Wt,
                          int K, int KP, int Ncols) {
    long long gid = (long long)blockIdx.x * blockDim.x + threadIdx.x;
    if (gid >= (long long)Ncols * KP) return;
    int c = (int)(gid / KP), k = (int)(gid % KP);
    Wt[gid] = (k < K) ? (_Float16)W[(long long)k * Ncols + c] : (_Float16)0.f;
}

__global__ void k_zero(unsigned* __restrict__ p, long long n) {
    long long gid = (long long)blockIdx.x * blockDim.x + threadIdx.x;
    if (gid < n) p[gid] = 0u;
}

// ---------------- WMMA GEMM: C[M,NT*16] = A[M,K] * B[K,NT*16] ----------------
// A: f16 row-major (lda=K, K%32==0).  Bt: f16 transposed [NT*16, K].  C: f32.
// One wave computes a 16-row strip across all NT column tiles (A-frag reuse).
// K is a compile-time constant so all B tile offsets are load immediates
// (single base pointer, no per-tile address VGPRs -> no spills), and the next
// B fragment is prefetched while the current WMMA executes.
template <int NT, int K>
__global__ void gemm_wmma(const _Float16* __restrict__ A,
                          const _Float16* __restrict__ Bt,
                          float* __restrict__ C) {
    int wave = (int)(((long long)blockIdx.x * blockDim.x + threadIdx.x) >> 5);
    int lane = threadIdx.x & 31;
    if (wave >= MTILES) return;                 // wave-uniform exit (EXEC all-ones)
    const int l15 = lane & 15;
    const int hi  = lane >> 4;                  // 0: lanes 0-15, 1: lanes 16-31
    const int row = wave * 16 + l15;
    const int ka  = hi * 16;                    // A k-offset (contiguous, matches B)
    const int kb  = hi * 16;                    // B k-offset (contiguous, matches A)
    const _Float16* arow  = A + (long long)row * K + ka;
    const _Float16* bbase = Bt + (long long)l15 * K + kb;

    v8f acc[NT];
#pragma unroll
    for (int t = 0; t < NT; ++t) { v8f z = {}; acc[t] = z; }

    union AU { v16h v; v8h h[2]; };

    for (int k0 = 0; k0 < K; k0 += 32) {
        AU au;
        au.h[0] = *(const v8h*)(arow);          // K = k0+ka .. +7
        au.h[1] = *(const v8h*)(arow + 8);      // K = k0+ka+8 .. +15
        v16h a = au.v;
        arow += 32;

        v16h bcur = *(const v16h*)(bbase);      // tile 0 (offset 0)
#pragma unroll
        for (int t = 0; t < NT - 1; ++t) {
            // compile-time immediate offset: next tile's fragment
            v16h bnext = *(const v16h*)(bbase + (long long)(t + 1) * 16 * K);
            acc[t] = __builtin_amdgcn_wmma_f32_16x16x32_f16(
                false, a, false, bcur, (short)0, acc[t], false, false);
            bcur = bnext;
        }
        acc[NT - 1] = __builtin_amdgcn_wmma_f32_16x16x32_f16(
            false, a, false, bcur, (short)0, acc[NT - 1], false, false);
        bbase += 32;
    }
    // C/D layout: vgpr i -> row = i + 8*hi, col = lane&15
    const int rbase = wave * 16 + hi * 8;
    const int ldc = NT * 16;
#pragma unroll
    for (int t = 0; t < NT; ++t) {
        int col = t * 16 + l15;
#pragma unroll
        for (int i = 0; i < 8; ++i)
            C[(long long)(rbase + i) * ldc + col] = acc[t][i];
    }
}

// ---------------- attention coefficients ----------------
__global__ void k_att(const float* __restrict__ h, const float* __restrict__ asrc,
                      const float* __restrict__ adst, float* __restrict__ als,
                      float* __restrict__ ald, int Cc) {
    long long gid = (long long)blockIdx.x * blockDim.x + threadIdx.x;
    if (gid >= (long long)N_NODES * 2) return;
    int n = (int)(gid >> 1), head = (int)(gid & 1);
    const float* hv = h + (long long)n * (2 * Cc) + head * Cc;
    const float* as_ = asrc + head * Cc;
    const float* ad_ = adst + head * Cc;
    float ss = 0.f, sd = 0.f;
    for (int c = 0; c < Cc; ++c) { float v = hv[c]; ss += v * as_[c]; sd += v * ad_[c]; }
    als[gid] = ss; ald[gid] = sd;
}

// ---------------- edge pass 1: per-dst max (for softmax stability) ----------------
__global__ void k_edge_max(const int* __restrict__ adj, const float* __restrict__ als,
                           const float* __restrict__ ald, unsigned* __restrict__ emax) {
    long long gid = (long long)blockIdx.x * blockDim.x + threadIdx.x;
    if (gid >= (long long)ETOT * 2) return;
    int eid = (int)(gid >> 1), head = (int)(gid & 1);
    int s, d;
    if (eid < NEDGE) { s = adj[eid]; d = adj[NEDGE + eid]; }
    else             { s = d = eid - NEDGE; }
    float e = lrelu_f(als[s * 2 + head] + ald[d * 2 + head]);
    atomicMax(&emax[d * 2 + head], fenc(e));
}

// ---------------- edge pass 2: exp, denominator, weighted scatter ----------------
// one wave per edge; lanes stride the HC channels
__global__ void k_edge_scatter(const int* __restrict__ adj, const float* __restrict__ als,
                               const float* __restrict__ ald, const unsigned* __restrict__ emax,
                               float* __restrict__ den, const float* __restrict__ h,
                               float* __restrict__ oacc, int Cc, int HC) {
    long long gid = (long long)blockIdx.x * blockDim.x + threadIdx.x;
    int eid = (int)(gid >> 5);
    int lane = (int)(gid & 31);
    if (eid >= ETOT) return;                    // wave-uniform
    int s, d;
    if (eid < NEDGE) { s = adj[eid]; d = adj[NEDGE + eid]; }
    else             { s = d = eid - NEDGE; }
    float ex0, ex1;
    {
        float e0 = lrelu_f(als[s * 2 + 0] + ald[d * 2 + 0]);
        float e1 = lrelu_f(als[s * 2 + 1] + ald[d * 2 + 1]);
        ex0 = __expf(e0 - fdec(emax[d * 2 + 0]));
        ex1 = __expf(e1 - fdec(emax[d * 2 + 1]));
    }
    if (lane == 0) {
        atomicAdd(&den[d * 2 + 0], ex0);
        atomicAdd(&den[d * 2 + 1], ex1);
    }
    const float* hs = h + (long long)s * HC;
    float* od = oacc + (long long)d * HC;
    for (int ch = lane; ch < HC; ch += 32) {
        float w = (ch < Cc) ? ex0 : ex1;
        atomicAdd(&od[ch], w * hs[ch]);
    }
}

// ---------------- finalize: normalize + bias + ELU ----------------
__global__ void k_finalize_concat(const float* __restrict__ oacc, const float* __restrict__ den,
                                  const float* __restrict__ b, float* __restrict__ st,
                                  int Cc, int HC) {
    long long gid = (long long)blockIdx.x * blockDim.x + threadIdx.x;
    if (gid >= (long long)N_NODES * HC) return;
    int n = (int)(gid / HC), ch = (int)(gid % HC);
    int head = ch / Cc;
    float v = oacc[gid] / (den[n * 2 + head] + 1e-16f) + b[ch];
    st[gid] = elu_f(v);
}

__global__ void k_finalize_mean(const float* __restrict__ oacc, const float* __restrict__ den,
                                const float* __restrict__ b, float* __restrict__ st) {
    long long gid = (long long)blockIdx.x * blockDim.x + threadIdx.x;
    if (gid >= (long long)N_NODES * C1) return;
    int n = (int)(gid / C1), c = (int)(gid % C1);
    float v0 = oacc[(long long)n * HC1 + c]      / (den[n * 2 + 0] + 1e-16f);
    float v1 = oacc[(long long)n * HC1 + C1 + c] / (den[n * 2 + 1] + 1e-16f);
    st[gid] = elu_f(0.5f * (v0 + v1) + b[c]);
}

// ---------------- NOrderAggr: elu(mix) then learned combine ----------------
__global__ void k_aggr(const float* __restrict__ st, const float* __restrict__ Wmix,
                       const float* __restrict__ wfin, int Cc,
                       _Float16* __restrict__ out_h, float* __restrict__ out_f) {
    long long gid = (long long)blockIdx.x * blockDim.x + threadIdx.x;
    if (gid >= (long long)N_NODES * Cc) return;
    float s0 = st[gid];
    float s1 = st[(long long)N_NODES * Cc + gid];
    float s2 = st[(long long)2 * N_NODES * Cc + gid];
    float acc = 0.f;
#pragma unroll
    for (int j = 0; j < 3; ++j) {
        float t = s0 * Wmix[0 * 3 + j] + s1 * Wmix[1 * 3 + j] + s2 * Wmix[2 * 3 + j];
        acc += elu_f(t) * wfin[j];
    }
    if (out_h) out_h[gid] = (_Float16)acc;
    else       out_f[gid] = acc;
}

// ---------------- log_softmax over 40 classes, one wave per node ----------------
__global__ void k_logsoftmax(const float* __restrict__ xin, float* __restrict__ out) {
    long long gid = (long long)blockIdx.x * blockDim.x + threadIdx.x;
    int node = (int)(gid >> 5);
    int lane = (int)(gid & 31);
    if (node >= N_NODES) return;                // wave-uniform
    const float* v = xin + (long long)node * C1;
    float v0 = v[lane];
    bool has2 = (lane + 32) < C1;
    float v1 = has2 ? v[lane + 32] : -3.0e38f;
    float m = fmaxf(v0, v1);
    for (int off = 16; off > 0; off >>= 1) m = fmaxf(m, __shfl_xor(m, off, 32));
    float s = __expf(v0 - m) + (has2 ? __expf(v1 - m) : 0.f);
    for (int off = 16; off > 0; off >>= 1) s += __shfl_xor(s, off, 32);
    float lg = m + __logf(s);
    float* o = out + (long long)node * C1;
    o[lane] = v0 - lg;
    if (has2) o[lane + 32] = v1 - lg;
}

extern "C" void kernel_launch(void* const* d_in, const int* in_sizes, int n_in,
                              void* d_out, int out_size, void* d_ws, size_t ws_size,
                              hipStream_t stream) {
    (void)in_sizes; (void)n_in; (void)out_size; (void)ws_size;
    const float* x = (const float*)d_in[0];
    const int* adj[3] = {(const int*)d_in[1], (const int*)d_in[2], (const int*)d_in[3]};
    const float *W0[3], *AS0[3], *AD0[3], *B0[3], *W1[3], *AS1[3], *AD1[3], *B1[3];
    for (int o = 0; o < 3; ++o) {
        W0[o]  = (const float*)d_in[4 + 4 * o];
        AS0[o] = (const float*)d_in[5 + 4 * o];
        AD0[o] = (const float*)d_in[6 + 4 * o];
        B0[o]  = (const float*)d_in[7 + 4 * o];
        W1[o]  = (const float*)d_in[16 + 4 * o];
        AS1[o] = (const float*)d_in[17 + 4 * o];
        AD1[o] = (const float*)d_in[18 + 4 * o];
        B1[o]  = (const float*)d_in[19 + 4 * o];
    }
    const float* aggW0 = (const float*)d_in[28];
    const float* aggw0 = (const float*)d_in[29];
    const float* aggW1 = (const float*)d_in[30];
    const float* aggw1 = (const float*)d_in[31];
    float* out = (float*)d_out;

    // workspace carve-up (256B aligned)
    char* ws = (char*)d_ws;
    size_t off = 0;
    auto alloc = [&](size_t sz) -> char* {
        char* p = ws + off;
        off = (off + sz + 255) & ~(size_t)255;
        return p;
    };
    _Float16* xh   = (_Float16*)alloc((size_t)N_NODES * KP0 * 2);     // padded f16 x
    _Float16* wt   = (_Float16*)alloc((size_t)KP0 * HC0 * 2);         // transposed f16 W
    float*    hbuf = (float*)alloc((size_t)N_NODES * HC0 * 4);        // GEMM output
    float*    als  = (float*)alloc((size_t)N_NODES * 2 * 4);
    float*    ald  = (float*)alloc((size_t)N_NODES * 2 * 4);
    unsigned* emax = (unsigned*)alloc((size_t)N_NODES * 2 * 4);
    float*    den  = (float*)alloc((size_t)N_NODES * 2 * 4);
    float*    oacc = (float*)alloc((size_t)N_NODES * HC0 * 4);
    float*    st0  = (float*)alloc((size_t)3 * N_NODES * HC0 * 4);
    float*    st1  = (float*)alloc((size_t)3 * N_NODES * C1 * 4);
    _Float16* x1h  = (_Float16*)alloc((size_t)N_NODES * KP1 * 2);
    float*    xfin = (float*)alloc((size_t)N_NODES * C1 * 4);

    const int T = 256;
    auto blk = [](long long n) { return (unsigned)((n + 255) / 256); };
    const unsigned gemm_blocks = (MTILES + 7) / 8;   // 8 waves per block

    k_conv_x<<<blk((long long)N_NODES * KP0), T, 0, stream>>>(x, xh);

    // ---- layer 0 (concat heads) ----
    for (int o = 0; o < 3; ++o) {
        k_conv_wt<<<blk((long long)HC0 * KP0), T, 0, stream>>>(W0[o], wt, F_IN, KP0, HC0);
        gemm_wmma<8, KP0><<<gemm_blocks, T, 0, stream>>>(xh, wt, hbuf);
        k_att<<<blk((long long)N_NODES * 2), T, 0, stream>>>(hbuf, AS0[o], AD0[o], als, ald, C0);
        long long zc = (((char*)(oacc + (size_t)N_NODES * HC0)) - (char*)emax) / 4;
        k_zero<<<blk(zc), T, 0, stream>>>(emax, zc);     // clears emax, den, oacc
        k_edge_max<<<blk((long long)ETOT * 2), T, 0, stream>>>(adj[o], als, ald, emax);
        k_edge_scatter<<<blk((long long)ETOT * 32), T, 0, stream>>>(
            adj[o], als, ald, emax, den, hbuf, oacc, C0, HC0);
        k_finalize_concat<<<blk((long long)N_NODES * HC0), T, 0, stream>>>(
            oacc, den, B0[o], st0 + (size_t)o * N_NODES * HC0, C0, HC0);
    }
    k_aggr<<<blk((long long)N_NODES * HC0), T, 0, stream>>>(st0, aggW0, aggw0, HC0, x1h, nullptr);

    // ---- layer 1 (mean heads) ----
    for (int o = 0; o < 3; ++o) {
        k_conv_wt<<<blk((long long)HC1 * KP1), T, 0, stream>>>(W1[o], wt, KP1, KP1, HC1);
        gemm_wmma<5, KP1><<<gemm_blocks, T, 0, stream>>>(x1h, wt, hbuf);
        k_att<<<blk((long long)N_NODES * 2), T, 0, stream>>>(hbuf, AS1[o], AD1[o], als, ald, C1);
        long long zc = (((char*)(oacc + (size_t)N_NODES * HC1)) - (char*)emax) / 4;
        k_zero<<<blk(zc), T, 0, stream>>>(emax, zc);
        k_edge_max<<<blk((long long)ETOT * 2), T, 0, stream>>>(adj[o], als, ald, emax);
        k_edge_scatter<<<blk((long long)ETOT * 32), T, 0, stream>>>(
            adj[o], als, ald, emax, den, hbuf, oacc, C1, HC1);
        k_finalize_mean<<<blk((long long)N_NODES * C1), T, 0, stream>>>(
            oacc, den, B1[o], st1 + (size_t)o * N_NODES * C1);
    }
    k_aggr<<<blk((long long)N_NODES * C1), T, 0, stream>>>(st1, aggW1, aggw1, C1, nullptr, xfin);

    k_logsoftmax<<<blk((long long)N_NODES * 32), T, 0, stream>>>(xfin, out);
}
